// EuclideanCodebook_618475291340
// MI455X (gfx1250) — matrix operand order repaired and checked by reading, not compiled, not run on hardware
//
#include <hip/hip_runtime.h>
#include <hip/hip_bf16.h>
#include <cstdint>

#define DIM        512
#define KCODES     4096
#define NROWS      32768
#define DECAY      0.8f
#define EPS        1e-5f

#define MT    128                 // rows per block
#define CT    64                  // codes per chunk
#define XSE   520                 // padded LDS row stride (bf16 elems) = 1040 B
#define XSB   1040                // padded LDS row stride (bytes)
#define GRB   1024                // global bf16 row stride (bytes), unpadded
#define LDSX_SZ  (MT * XSB)       // 133120
#define LDSE_SZ  (CT * XSB)       //  66560
#define RED_OFF  (LDSX_SZ + 2 * LDSE_SZ)          // 266240
#define SHMEM_SZ (RED_OFF + MT * 16 * 8)          // 282624

typedef __attribute__((ext_vector_type(16))) __bf16 v16bf;
typedef __attribute__((ext_vector_type(8)))  float  v8f;

// f32 -> bf16 RNE via integer ops
__device__ inline unsigned short f2bfu(float f) {
    union { float f; unsigned u; } a; a.f = f;
    unsigned r = a.u + 0x7FFFu + ((a.u >> 16) & 1u);
    return (unsigned short)(r >> 16);
}
__device__ inline unsigned pack2(float lo, float hi) {
    return (unsigned)f2bfu(lo) | ((unsigned)f2bfu(hi) << 16);
}

union FragBF { uint4 q[2]; v16bf v; };

// ---------------------------------------- embed: bf16 copy + ||e_k||^2
__global__ __launch_bounds__(256)
void vq_prep_embed(const float* __restrict__ embed, unsigned* __restrict__ ebf,
                   float* __restrict__ e2) {
    __shared__ float red[256];
    int k = blockIdx.x, t = threadIdx.x;
    float2 v = ((const float2*)embed)[(size_t)k * 256 + t];
    ebf[(size_t)k * 256 + t] = pack2(v.x, v.y);
    red[t] = v.x * v.x + v.y * v.y; __syncthreads();
    for (int o = 128; o > 0; o >>= 1) { if (t < o) red[t] += red[t + o]; __syncthreads(); }
    if (!t) e2[k] = red[0];
}

// ---------------------------------------- x: bf16 copy (one streaming pass)
__global__ __launch_bounds__(256)
void vq_prep_x(const float* __restrict__ x, unsigned* __restrict__ xbf) {
    size_t i = (size_t)blockIdx.x * 256 + threadIdx.x;     // over NROWS*DIM/4
    float4 v = ((const float4*)x)[i];
    uint2 o; o.x = pack2(v.x, v.y); o.y = pack2(v.z, v.w);
    ((uint2*)xbf)[i] = o;
}

// ---- async global->LDS tile movers (bf16 payload, 16B chunks, padded LDS rows)
__device__ inline void issue_async_x(unsigned ldsDst, unsigned long long gbase,
                                     int R, int tid) {
#pragma unroll
    for (int it = 0; it < 32; ++it) {                      // 128 rows x 64 chunks
        int g = tid + it * 256;
        int row = g >> 6, c = g & 63;
        unsigned goff = (unsigned)((R + row) * GRB + c * 16);
        unsigned ldst = ldsDst + (unsigned)(row * XSB + c * 16);
        asm volatile("global_load_async_to_lds_b128 %0, %1, %2"
                     :: "v"(ldst), "v"(goff), "s"(gbase) : "memory");
    }
}
__device__ inline void issue_async_e(unsigned ldsDst, unsigned long long gbase,
                                     int C0, int tid) {
#pragma unroll
    for (int it = 0; it < 16; ++it) {                      // 64 rows x 64 chunks
        int g = tid + it * 256;
        int row = g >> 6, c = g & 63;
        unsigned goff = (unsigned)((C0 + row) * GRB + c * 16);
        unsigned ldst = ldsDst + (unsigned)(row * XSB + c * 16);
        asm volatile("global_load_async_to_lds_b128 %0, %1, %2"
                     :: "v"(ldst), "v"(goff), "s"(gbase) : "memory");
    }
}

// ------------------------------------------- WMMA bf16 GEMM + running argmax
__global__ __launch_bounds__(256)
void vq_argmax_wmma(const unsigned* __restrict__ xbf, const unsigned* __restrict__ ebf,
                    const float* __restrict__ e2, int* __restrict__ ind) {
    extern __shared__ char smem[];
    float* redV = (float*)(smem + RED_OFF);
    int*   redI = (int*)(smem + RED_OFF + MT * 16 * 4);

    const int tid = threadIdx.x;
    const int wave = tid >> 5, lane = tid & 31;
    const int lane16 = lane & 15, laneHalf = lane >> 4;
    const int R = blockIdx.x * MT;
    const unsigned ldsBase = (unsigned)(uintptr_t)smem;    // low 32 bits = LDS offset
    const unsigned long long xbase = (unsigned long long)(uintptr_t)xbf;
    const unsigned long long ebase = (unsigned long long)(uintptr_t)ebf;

    // prologue: x tile (once) + embed chunk 0 into buffer 0
    issue_async_x(ldsBase, xbase, R, tid);
    issue_async_e(ldsBase + LDSX_SZ, ebase, 0, tid);

    const int ms = wave * 16;                              // this wave's row strip
    float best[8]; int bidx[8];
#pragma unroll
    for (int j = 0; j < 8; ++j) { best[j] = -3.0e38f; bidx[j] = 0; }
    const v8f vzero = {0.f,0.f,0.f,0.f,0.f,0.f,0.f,0.f};

    for (int c = 0; c < KCODES / CT; ++c) {
        const int C0 = c * CT;
        __syncthreads();                                   // prev compute done -> buf reusable
        if (c + 1 < KCODES / CT) {
            issue_async_e(ldsBase + LDSX_SZ + (unsigned)(((c + 1) & 1) * LDSE_SZ),
                          ebase, C0 + CT, tid);
            asm volatile("s_wait_asynccnt 16" ::: "memory");   // chunk c (and x) landed
        } else {
            asm volatile("s_wait_asynccnt 0" ::: "memory");
        }
        __syncthreads();                                   // all waves' copies visible

        const __bf16* xb = (const __bf16*)smem;
        const __bf16* eb = (const __bf16*)(smem + LDSX_SZ + (size_t)(c & 1) * LDSE_SZ);

        v8f acc[4];
#pragma unroll
        for (int tb = 0; tb < 4; ++tb) acc[tb] = vzero;

        const int arow = ms + lane16;
#pragma unroll 4
        for (int kk = 0; kk < DIM / 32; ++kk) {
            const int kb = kk * 32;
            FragBF a;
            // A 16x32 bf16: lanes0-15 K={0..7,16..23}, lanes16-31 K={8..15,24..31}
            a.q[0] = *(const uint4*)(xb + arow * XSE + kb + laneHalf * 8);
            a.q[1] = *(const uint4*)(xb + arow * XSE + kb + 16 + laneHalf * 8);
#pragma unroll
            for (int tb = 0; tb < 4; ++tb) {
                // B 32x16 bf16: lane = column(code); lanes0-15 K={0..7,16..23}, lanes16-31 K={8..15,24..31}
                const __bf16* bp = eb + (tb * 16 + lane16) * XSE + kb + laneHalf * 8;
                FragBF b;
                b.q[0] = *(const uint4*)(bp);
                b.q[1] = *(const uint4*)(bp + 16);
                acc[tb] = __builtin_amdgcn_wmma_f32_16x16x32_bf16(
                              false, a.v, false, b.v, (short)0, acc[tb], false, false);
            }
        }

        // score = 2*(x.e) - ||e||^2 ; fold into running argmax
#pragma unroll
        for (int tb = 0; tb < 4; ++tb) {
            int col = C0 + tb * 16 + lane16;
            float e2v = e2[col];
#pragma unroll
            for (int j = 0; j < 8; ++j) {
                float s = 2.f * acc[tb][j] - e2v;
                if (s > best[j]) { best[j] = s; bidx[j] = col; }
            }
        }
    }

    // per-row reduction: each row owned by one wave -> 16 lane candidates
#pragma unroll
    for (int j = 0; j < 8; ++j) {
        int rloc = ms + j + 8 * laneHalf;
        redV[rloc * 16 + lane16] = best[j];
        redI[rloc * 16 + lane16] = bidx[j];
    }
    __syncthreads();
    if (tid < MT) {
        float bv = -3.0e38f; int bi = 0;
        for (int s = 0; s < 16; ++s) {
            float v = redV[tid * 16 + s];
            if (v > bv) { bv = v; bi = redI[tid * 16 + s]; }
        }
        ind[R + tid] = bi;
    }
}

// -------------------------------------------------- quantize gather + ind out
__global__ __launch_bounds__(256)
void vq_gather(const float* __restrict__ embed, const int* __restrict__ ind,
               float* __restrict__ outq, float* __restrict__ outind) {
    size_t i = (size_t)blockIdx.x * 256 + threadIdx.x;
    int n = (int)(i >> 9), d = (int)(i & 511);
    int c = ind[n];
    outq[i] = embed[(size_t)c * DIM + d];
    if (d == 0) outind[n] = (float)c;
}

// ------------- deterministic segment sums: tiled match-compaction per code
#define TILE_N 4096
__global__ __launch_bounds__(256)
void vq_segsum(const float* __restrict__ x, const int* __restrict__ ind,
               float* __restrict__ bcs, float* __restrict__ esum) {
    __shared__ int cnts[256];
    __shared__ int list[TILE_N];
    int k = blockIdx.x, t = threadIdx.x;
    float a0 = 0.f, a1 = 0.f;
    int total = 0;
    for (int tile = 0; tile < NROWS; tile += TILE_N) {
        int base = tile + t * 16;
        int local[16]; int cnt = 0;
#pragma unroll
        for (int i = 0; i < 16; ++i)
            if (ind[base + i] == k) local[cnt++] = base + i;
        cnts[t] = cnt;
        __syncthreads();
        for (int d = 1; d < 256; d <<= 1) {          // inclusive scan of counts
            int add = (t >= d) ? cnts[t - d] : 0;
            __syncthreads();
            cnts[t] += add;
            __syncthreads();
        }
        int excl = cnts[t] - cnt;
        int tot = cnts[255];
        for (int i = 0; i < cnt; ++i) list[excl + i] = local[i];  // ascending rows
        __syncthreads();
        for (int m = 0; m < tot; ++m) {              // fixed order -> deterministic
            const float* xr = x + (size_t)list[m] * DIM;
            a0 += xr[t]; a1 += xr[t + 256];
        }
        total += tot;
        __syncthreads();
    }
    esum[(size_t)k * DIM + t] = a0;
    esum[(size_t)k * DIM + t + 256] = a1;
    if (!t) bcs[k] = (float)total;
}

// --------------------- EMA counts, perplexity, Laplace-smoothed divisors
__global__ __launch_bounds__(1024)
void vq_stats(const float* __restrict__ cs, const float* __restrict__ bcs,
              float* __restrict__ out_ncs, float* __restrict__ out_perp,
              float* __restrict__ smoothed) {
    __shared__ float sT[1024], sH[1024];
    int t = threadIdx.x;
    float tot = 0.f, h = 0.f, ncsv[4];
#pragma unroll
    for (int j = 0; j < 4; ++j) {
        int k = t + j * 1024;
        float v = cs[k] * DECAY + bcs[k] * (1.f - DECAY);
        out_ncs[k] = v; ncsv[j] = v;
        tot += v; h += v * logf(v + EPS);
    }
    sT[t] = tot; sH[t] = h; __syncthreads();
    for (int o = 512; o > 0; o >>= 1) {
        if (t < o) { sT[t] += sT[t + o]; sH[t] += sH[t + o]; }
        __syncthreads();
    }
    float total = sT[0];
    if (!t) out_perp[0] = expf(-sH[0]);
    float scale = total / (total + (float)KCODES * EPS);
#pragma unroll
    for (int j = 0; j < 4; ++j) smoothed[t + j * 1024] = (ncsv[j] + EPS) * scale;
}

// ----------------------------------- EMA embed_avg + normalized new_embed
__global__ __launch_bounds__(256)
void vq_embed_update(const float* __restrict__ ea, const float* __restrict__ es,
                     const float* __restrict__ smoothed,
                     float* __restrict__ out_nea, float* __restrict__ out_ne) {
    size_t i = (size_t)blockIdx.x * 256 + threadIdx.x;
    int k = (int)(i >> 9);
    float v = ea[i] * DECAY + es[i] * (1.f - DECAY);
    out_nea[i] = v;
    out_ne[i] = v / smoothed[k];
}

extern "C" void kernel_launch(void* const* d_in, const int* in_sizes, int n_in,
                              void* d_out, int out_size, void* d_ws, size_t ws_size,
                              hipStream_t stream) {
    const float* x     = (const float*)d_in[0];   // [8,4096,512]
    const float* embed = (const float*)d_in[1];   // [4096,512]
    const float* cs    = (const float*)d_in[2];   // [4096]
    const float* ea    = (const float*)d_in[3];   // [4096,512]

    float* out  = (float*)d_out;
    float* OQ   = out;                            // quantize       16777216
    float* OI   = OQ + 16777216;                  // embed_ind        32768
    float* OP   = OI + 32768;                     // perplexity           1
    float* ONE  = OP + 1;                         // new_embed      2097152
    float* ONCS = ONE + 2097152;                  // new_cluster      4096
    float* ONEA = ONCS + 4096;                    // new_embed_avg  2097152

    char* ws = (char*)d_ws;
    float*    e2       = (float*)   (ws);               //  16 KB
    int*      ind      = (int*)     (ws + 16384);       // 128 KB
    float*    bcs      = (float*)   (ws + 147456);      //  16 KB
    float*    esum     = (float*)   (ws + 163840);      //   8 MB
    float*    smoothed = (float*)   (ws + 8552448);     //  16 KB
    unsigned* xbf      = (unsigned*)(ws + 8568832);     //  32 MB (bf16 x)
    unsigned* ebf      = (unsigned*)(ws + 42123264);    //   4 MB (bf16 embed)

    vq_prep_embed<<<KCODES, 256, 0, stream>>>(embed, ebf, e2);
    vq_prep_x<<<(NROWS * DIM / 4) / 256, 256, 0, stream>>>(x, xbf);

    vq_argmax_wmma<<<NROWS / MT, 256, SHMEM_SZ, stream>>>(xbf, ebf, e2, ind);

    vq_gather<<<(NROWS * DIM) / 256, 256, 0, stream>>>(embed, ind, OQ, OI);
    vq_segsum<<<KCODES, 256, 0, stream>>>(x, ind, bcs, esum);
    vq_stats<<<1, 1024, 0, stream>>>(cs, bcs, ONCS, OP, smoothed);
    vq_embed_update<<<(KCODES * DIM) / 256, 256, 0, stream>>>(ea, esum, smoothed, ONEA, ONE);
}
